// actor_65601330479204
// MI455X (gfx1250) — matrix operand order repaired and checked by reading, not compiled, not run on hardware
//
#include <hip/hip_runtime.h>
#include <hip/hip_bf16.h>
#include <math.h>

typedef __attribute__((ext_vector_type(16))) _Float16 v16h;
typedef __attribute__((ext_vector_type(8)))  float    v8f;
typedef __attribute__((ext_vector_type(4)))  float    v4f;

#define ZONES 128
#define NB    4096
#define NDEG  16
#define NE    (ZONES * NDEG)   // 2048 edges
#define BT    16               // batch tiles per wave (16 rows each)

// Branch-free fast activations: one v_exp_f32 + one v_rcp_f32 each.
__device__ __forceinline__ float fast_sigmoid(float x) {
  return __builtin_amdgcn_rcpf(1.f + __expf(-x));
}
__device__ __forceinline__ float fast_tanh(float x) {
  return 1.f - 2.f * __builtin_amdgcn_rcpf(1.f + __expf(2.f * x));
}

// ---------------------------------------------------------------------------
// Kernel 1: stream 1.0f into the whole [B, Z, Z] output with NT stores.
// 256 MB of writes -> the bandwidth floor (~11 us at 23.3 TB/s). NT keeps the
// 192 MB L2 free for x + weights (~2.3 MB) and the beta scatter.
// ---------------------------------------------------------------------------
__global__ __launch_bounds__(256) void fill_ones_kernel(float* __restrict__ out, int n4) {
  int i = blockIdx.x * blockDim.x + threadIdx.x;
  if (i < n4) {
    v4f v = {1.f, 1.f, 1.f, 1.f};
    __builtin_nontemporal_store(v, (v4f*)(out + (size_t)i * 4));
  }
}

// ---------------------------------------------------------------------------
// Kernel 2: per-edge tiny MLP batched onto WMMA.
// One wave owns 4 edges x 256 batch rows, processed as 16 tiles of
// (16 batch x 4 edges). The three block-diagonal f16 B matrices (W1/W2/W3)
// and all biases are built once and stay in registers across the batch loop.
//
// A f16 16x32: lane L<16 -> M=L, halves[0..7]=K0..7, [8..15]=K16..23;
//              lane L+16 -> M=L, halves[0..7]=K8..15, [8..15]=K24..31.
// B f16 32x16: lane n<16 -> N=n, halves[0..7]=K0..7, [8..15]=K16..23; lanes 16..31 -> 0.
// C/D f32 16x16: lane L<16 -> N=L, VGPR r = M=r; lane L+16 -> N=L-16, M=8+r.
// ---------------------------------------------------------------------------
__global__ __launch_bounds__(256) void edge_mlp_wmma_kernel(
    const float* __restrict__ x,
    const int*   __restrict__ src, const int* __restrict__ dst,
    const float* __restrict__ W1,  const float* __restrict__ b1,
    const float* __restrict__ W2,  const float* __restrict__ b2,
    const float* __restrict__ W3,  const float* __restrict__ b3,
    float* __restrict__ out)
{
  __shared__ float Tall[8 * 256];              // one 16x16 f32 tile per wave
  const int lane = threadIdx.x & 31;
  const int wib  = threadIdx.x >> 5;
  float* T = Tall + wib * 256;

  const int gwave   = blockIdx.x * (blockDim.x >> 5) + wib;
  const int TILES_E = NE / 4;                  // 512
  const int te  = gwave % TILES_E;
  const int tbO = gwave / TILES_E;             // 0..15 (outer batch tile)
  const int e0  = te * 4;                      // first of 4 edges for this wave

  const int n  = lane & 15;                    // column of the 16x16 D tile
  const int j  = n >> 2;                       // edge slot of this column
  const int o  = n & 3;                        // output feature of this column
  const int en = e0 + j;                       // edge id for this lane's column
  const int rbase = (lane < 16) ? 0 : 8;       // D rows held by this lane
  const int koff  = (lane < 16) ? 0 : 8;       // A K-offset for transposed read
  const int M     = lane & 15;                 // A row for this lane

  // ---- hoisted, loop-invariant state --------------------------------------
  // src/dst columns for the 4 edges (used by lanes<16 for the x gather)
  int sc[4], dc[4];
  #pragma unroll
  for (int jj = 0; jj < 4; ++jj) { sc[jj] = src[e0 + jj]; dc[jj] = dst[e0 + jj]; }

  // block-diagonal B matrices for the three layers
  v16h bm1, bm2, bm3;
  #pragma unroll
  for (int k = 0; k < 16; ++k) {
    bm1[k] = (_Float16)0.f; bm2[k] = (_Float16)0.f; bm3[k] = (_Float16)0.f;
  }
  if (lane < 16) {
    bm1[2 * j]     = (_Float16)W1[en * 8 + 0 * 4 + o];   // W1 (E,2,4)
    bm1[2 * j + 1] = (_Float16)W1[en * 8 + 1 * 4 + o];
    #pragma unroll
    for (int i = 0; i < 4; ++i)
      bm2[4 * j + i] = (_Float16)W2[en * 16 + i * 4 + o]; // W2 (E,4,4)
  }
  if (lane < 4) {
    #pragma unroll
    for (int oo = 0; oo < 4; ++oo)
      bm3[4 * lane + oo] = (_Float16)W3[(e0 + lane) * 4 + oo]; // W3 (E,4)
  }

  const float bias1 = b1[en * 4 + o];
  const float bias2 = b2[en * 4 + o];
  // scatter target for columns 0..3
  const int   eout = e0 + n;                   // valid when n < 4
  const float b3v  = (n < 4) ? b3[eout] : 0.f;
  const size_t colofs = (n < 4)
      ? ((size_t)src[eout] * ZONES + (size_t)dst[eout]) : 0;

  v16h a;
  v8f  acc;

  // ---- batch loop: 16 tiles of 16 rows ------------------------------------
  for (int it = 0; it < BT; ++it) {
    const int b0 = tbO * (16 * BT) + it * 16;

    // layer 1: A = per-edge [x_src, x_dst] pairs in K slots 2j..2j+1
    #pragma unroll
    for (int k = 0; k < 16; ++k) a[k] = (_Float16)0.f;
    if (lane < 16) {
      const float* xrow = x + (size_t)(b0 + lane) * ZONES;
      #pragma unroll
      for (int jj = 0; jj < 4; ++jj) {
        a[2 * jj]     = (_Float16)xrow[sc[jj]];
        a[2 * jj + 1] = (_Float16)xrow[dc[jj]];
      }
    }
    #pragma unroll
    for (int r = 0; r < 8; ++r) acc[r] = 0.f;
    acc = __builtin_amdgcn_wmma_f32_16x16x32_f16(false, a, false, bm1,
                                                 (short)0, acc, false, false);
    #pragma unroll
    for (int r = 0; r < 8; ++r)
      T[(rbase + r) * 16 + n] = fast_tanh(acc[r] + bias1);
    __syncthreads();

    // layer 2: transpose h1 out of LDS into A, block-diagonal W2
    #pragma unroll
    for (int k = 0; k < 16; ++k) a[k] = (lane < 16) ? (_Float16)T[M * 16 + k] : (_Float16)0.f;
    #pragma unroll
    for (int r = 0; r < 8; ++r) acc[r] = 0.f;
    acc = __builtin_amdgcn_wmma_f32_16x16x32_f16(false, a, false, bm2,
                                                 (short)0, acc, false, false);
    __syncthreads();                           // done reading T
    #pragma unroll
    for (int r = 0; r < 8; ++r)
      T[(rbase + r) * 16 + n] = fast_tanh(acc[r] + bias2);
    __syncthreads();

    // layer 3: dot(h2, W3) per edge -> columns 0..3 of D
    #pragma unroll
    for (int k = 0; k < 16; ++k) a[k] = (lane < 16) ? (_Float16)T[M * 16 + k] : (_Float16)0.f;
    #pragma unroll
    for (int r = 0; r < 8; ++r) acc[r] = 0.f;
    acc = __builtin_amdgcn_wmma_f32_16x16x32_f16(false, a, false, bm3,
                                                 (short)0, acc, false, false);

    // sigmoid + scatter: 4 edges x 8 rows per active lane pair
    if (n < 4) {
      size_t base = (size_t)(b0 + rbase) * (ZONES * ZONES) + colofs;
      #pragma unroll
      for (int r = 0; r < 8; ++r)
        out[base + (size_t)r * (ZONES * ZONES)] = fast_sigmoid(acc[r] + b3v);
    }
    __syncthreads();                           // done reading T this iteration
  }
}

// ---------------------------------------------------------------------------
extern "C" void kernel_launch(void* const* d_in, const int* in_sizes, int n_in,
                              void* d_out, int out_size, void* d_ws, size_t ws_size,
                              hipStream_t stream) {
  const float* x   = (const float*)d_in[0];
  // d_in[1] = dtPt scalar (== NB), compile-time constant here
  const int*   src = (const int*)d_in[2];
  const int*   dst = (const int*)d_in[3];
  const float* W1  = (const float*)d_in[4];
  const float* b1  = (const float*)d_in[5];
  const float* W2  = (const float*)d_in[6];
  const float* b2  = (const float*)d_in[7];
  const float* W3  = (const float*)d_in[8];
  const float* b3  = (const float*)d_in[9];
  float* out = (float*)d_out;

  // 1) ones fill: 67,108,864 floats as 16B NT stores
  const int n4 = (NB * ZONES * ZONES) / 4;     // 16,777,216
  fill_ones_kernel<<<n4 / 256, 256, 0, stream>>>(out, n4);

  // 2) WMMA edge-MLP + scatter: (B/256)*(E/4) = 8192 waves, 8 waves/block
  const int waves = (NB / (16 * BT)) * (NE / 4);
  edge_mlp_wmma_kernel<<<waves / 8, 256, 0, stream>>>(
      x, src, dst, W1, b1, W2, b2, W3, b3, out);
}
